// MultiHeadSelfAttention_43233140801691
// MI455X (gfx1250) — compile-verified, matrix operand order checked by reading, not run on hardware
//
#include <hip/hip_runtime.h>
#include <hip/hip_bf16.h>

// ---------------------------------------------------------------------------
// Multi-head self-attention, CDNA5 (gfx1250), bf16 WMMA + TDM-staged GEMMs
// + register-resident flash attention.
//   B=2, T=2048, D=1024, H=16, HD=64
// ---------------------------------------------------------------------------

typedef __attribute__((ext_vector_type(16))) __bf16 v16bf;
typedef __attribute__((ext_vector_type(8)))  __bf16 v8bf;
typedef __attribute__((ext_vector_type(8)))  float  v8f;
typedef __attribute__((ext_vector_type(4)))  unsigned int u32x4;
typedef __attribute__((ext_vector_type(8)))  int i32x8;
typedef __attribute__((ext_vector_type(4)))  int i32x4;

#define WMMA_BF16(a, b, c) \
  __builtin_amdgcn_wmma_f32_16x16x32_bf16(false, (a), false, (b), (short)0, (c), false, false)

constexpr int Bn    = 2;
constexpr int T     = 2048;
constexpr int D     = 1024;
constexpr int H     = 16;
constexpr int HD    = 64;
constexpr int Mrows = Bn * T;     // 4096
constexpr int Nqkv  = 3 * D;      // 3072

// ---------------------------------------------------------------------------
// Tensor Data Mover: 2-D tile load (bf16 elements) into LDS.
// Builds a D# per CDNA5 ISA:
//   group0: count=1 | lds_addr | global_addr[56:0] | type=2
//   group1: data_size=1 (2 bytes), tensor_dim0/1, tile_dim0/1, dim0_stride
// 2-D tensors: groups 2/3 unused (zero).  6-arg builtin form (clang-23).
// ---------------------------------------------------------------------------
__device__ __forceinline__ void tdm_load_2d_bf16(unsigned lds_off,
                                                 const __bf16* gbase,
                                                 unsigned long long elem_off,
                                                 unsigned tensor_d0,  // row length (elems)
                                                 unsigned tensor_d1,  // #rows
                                                 unsigned tile_d0,    // tile cols (elems)
                                                 unsigned tile_d1,    // tile rows
                                                 unsigned stride_d0)  // row stride (elems)
{
  unsigned long long ga = (unsigned long long)(size_t)gbase + elem_off * 2ull;
  u32x4 g0;
  g0[0] = 1u;                                             // count=1, user mode
  g0[1] = lds_off;                                        // lds_addr (bytes)
  g0[2] = (unsigned)(ga & 0xffffffffull);                 // global_addr[31:0]
  g0[3] = (unsigned)((ga >> 32) & 0x01ffffffull) | (2u << 30);  // [56:32] | type=2
  i32x8 g1;
  g1[0] = (int)(1u << 16);                                // data_size=1 -> 2 bytes
  g1[1] = (int)((tensor_d0 & 0xffffu) << 16);             // tensor_dim0[15:0]
  g1[2] = (int)((tensor_d0 >> 16) | ((tensor_d1 & 0xffffu) << 16));
  g1[3] = (int)((tensor_d1 >> 16) | (tile_d0 << 16));     // dim1 hi | tile_dim0
  g1[4] = (int)(tile_d1 & 0xffffu);                       // tile_dim1 (tile_dim2=0)
  g1[5] = (int)stride_d0;                                 // tensor_dim0_stride[31:0]
  g1[6] = 0;                                              // stride hi, dim1_stride lo
  g1[7] = 0;
  i32x4 z4 = {0, 0, 0, 0};
  i32x8 z8 = {0, 0, 0, 0, 0, 0, 0, 0};
  __builtin_amdgcn_tensor_load_to_lds(g0, g1, z4, z4, z8, 0);
}

// ---------------------------------------------------------------------------
// Fragment loader: row-major 16-bit matrix, CDNA5 WMMA per-lane layout.
// Lane l (l<16) holds row `row`, K = k0 + {0..7, 16..23};
// lane l (l>=16) holds K = k0 + {8..15, 24..31}  (pass hi = lane>>4).
// Works for global or LDS base (addrspace inferred after inlining).
// ---------------------------------------------------------------------------
__device__ __forceinline__ v16bf load_frag_rm(const __bf16* __restrict__ base,
                                              int row, int ld, int k0, int hi) {
  const __bf16* p = base + (size_t)row * ld + k0 + hi * 8;
  v8bf lo = *(const v8bf*)(p);        // K = +0..7   (16B aligned)
  v8bf hh = *(const v8bf*)(p + 16);   // K = +16..23 (16B aligned)
  v16bf f;
#pragma unroll
  for (int i = 0; i < 8; ++i) { f[i] = lo[i]; f[8 + i] = hh[i]; }
  return f;
}

// ---------------------------------------------------------------------------
// Stage 0: f32 -> bf16 elementwise conversion
// ---------------------------------------------------------------------------
__global__ __launch_bounds__(256) void cvt_f32_bf16(const float* __restrict__ in,
                                                    __bf16* __restrict__ out, int n) {
  int i = blockIdx.x * blockDim.x + threadIdx.x;
  if (i < n) out[i] = (__bf16)in[i];
}

// ---------------------------------------------------------------------------
// Stage 1: QKV projection. C[m,n] = sum_k X[m,k] * Wqkv[n,k] + b[n]
// Block macro-tile 128(M) x 64(N), K staged in 64-wide steps via TDM.
// 8 waves arranged 4(M) x 2(N); each wave computes a 2x2 grid of 16x16 tiles.
// Epilogue: n = h*192 + j ; j<64 -> Q, j<128 -> K, else V (stored transposed).
// ---------------------------------------------------------------------------
__global__ __launch_bounds__(256) void qkv_gemm(const __bf16* __restrict__ X,
                                                const __bf16* __restrict__ W,
                                                const float* __restrict__ bias,
                                                __bf16* __restrict__ Q,
                                                __bf16* __restrict__ K,
                                                __bf16* __restrict__ VT) {
  __shared__ __bf16 sA[128 * 64];   // 16 KiB
  __shared__ __bf16 sW[64 * 64];    //  8 KiB

  const int wave = threadIdx.x >> 5;
  const int lane = threadIdx.x & 31;
  const int hi   = lane >> 4;
  const int ln   = lane & 15;

  const int nblk = Nqkv / 64;               // 48
  const int m0   = (blockIdx.x / nblk) * 128;
  const int n0   = (blockIdx.x % nblk) * 64;
  const int wm   = (wave >> 1) * 32;        // 0,32,64,96
  const int wn   = (wave & 1) * 32;         // 0,32

  const unsigned ldsA = (unsigned)(size_t)sA;
  const unsigned ldsW = (unsigned)(size_t)sW;

  v8f acc[2][2] = {};
  for (int k0 = 0; k0 < D; k0 += 64) {
    if (wave == 0) {
      tdm_load_2d_bf16(ldsA, X, (unsigned long long)m0 * D + k0,
                       D, Mrows, 64, 128, D);
      tdm_load_2d_bf16(ldsW, W, (unsigned long long)n0 * D + k0,
                       D, Nqkv, 64, 64, D);
      __builtin_amdgcn_s_wait_tensorcnt(0);
    }
    __syncthreads();
#pragma unroll
    for (int kk = 0; kk < 64; kk += 32) {
      v16bf a0 = load_frag_rm(sA, wm + ln,      64, kk, hi);
      v16bf a1 = load_frag_rm(sA, wm + 16 + ln, 64, kk, hi);
      v16bf b0 = load_frag_rm(sW, wn + ln,      64, kk, hi);
      v16bf b1 = load_frag_rm(sW, wn + 16 + ln, 64, kk, hi);
      acc[0][0] = WMMA_BF16(a0, b0, acc[0][0]);
      acc[0][1] = WMMA_BF16(a0, b1, acc[0][1]);
      acc[1][0] = WMMA_BF16(a1, b0, acc[1][0]);
      acc[1][1] = WMMA_BF16(a1, b1, acc[1][1]);
    }
    __syncthreads();
  }

  // D layout per tile: lane -> n = ln, reg r -> m = r + 8*hi
#pragma unroll
  for (int tj = 0; tj < 2; ++tj) {
    const int n     = n0 + wn + tj * 16 + ln;   // output channel in [0,3072)
    const int h     = n / (3 * HD);
    const int j     = n % (3 * HD);
    const int which = j / HD;
    const int d     = j % HD;
    const float bv  = bias[n];
#pragma unroll
    for (int ti = 0; ti < 2; ++ti) {
#pragma unroll
      for (int r = 0; r < 8; ++r) {
        const int m  = m0 + wm + ti * 16 + hi * 8 + r;   // global row = b*T + t
        const int bb = m / T;
        const int t  = m % T;
        const __bf16 hv = (__bf16)(acc[ti][tj][r] + bv);
        const int bh = bb * H + h;
        if (which == 0)      Q[((size_t)bh * T + t) * HD + d]  = hv;
        else if (which == 1) K[((size_t)bh * T + t) * HD + d]  = hv;
        else                 VT[((size_t)bh * HD + d) * T + t] = hv;  // transposed
      }
    }
  }
}

// ---------------------------------------------------------------------------
// Stage 2: flash attention. One wave handles 16 queries of one (b,h).
// S^T = K * Q^T  -> lane owns a query column; softmax stats per lane.
// O^T = V^T * P  -> probability regs are already in B-fragment layout.
// ---------------------------------------------------------------------------
__global__ __launch_bounds__(256) void attn_kernel(const __bf16* __restrict__ Q,
                                                   const __bf16* __restrict__ K,
                                                   const __bf16* __restrict__ VT,
                                                   __bf16* __restrict__ Aout) {
  const int wave = threadIdx.x >> 5;
  const int lane = threadIdx.x & 31;
  const int hi   = lane >> 4;
  const int ln   = lane & 15;

  const int bh   = blockIdx.x / (T / 128);
  const int qblk = blockIdx.x % (T / 128);
  const int b    = bh / H;
  const int h    = bh % H;
  const int q0   = qblk * 128 + wave * 16;

  const __bf16* Qb = Q  + (size_t)bh * T * HD;
  const __bf16* Kb = K  + (size_t)bh * T * HD;
  const __bf16* Vb = VT + (size_t)bh * HD * T;   // [64, 2048] row-major

  // Q^T B-fragments (reused across all key blocks); lane owns query q0+ln.
  const v16bf bq0 = load_frag_rm(Qb, q0 + ln, HD, 0,  hi);
  const v16bf bq1 = load_frag_rm(Qb, q0 + ln, HD, 32, hi);

  v8f o0 = {}, o1 = {}, o2 = {}, o3 = {};
  float m_st = -INFINITY, l_st = 0.f;
  const int   q     = q0 + ln;
  const float scale = 0.125f;              // 1/sqrt(64)
  const int   nkb   = (q0 + 47) >> 5;      // covers keys 0 .. q0+15 (causal)

  for (int kb = 0; kb < nkb; ++kb) {
    const int kbase = kb * 32;

    if (kb + 1 < nkb) {                    // prefetch next K/V blocks
      __builtin_prefetch(Kb + (size_t)(kbase + 32) * HD, 0, 0);
      __builtin_prefetch(Vb + (size_t)ln * T + kbase + 32, 0, 0);
    }

    // S^T tiles: m = key, n = query. Two 16-key subtiles, K-dim = 64.
    v8f s0 = {}, s1 = {};
    {
      v16bf ak0 = load_frag_rm(Kb, kbase + ln,      HD, 0,  hi);
      v16bf ak1 = load_frag_rm(Kb, kbase + ln,      HD, 32, hi);
      s0 = WMMA_BF16(ak0, bq0, s0);
      s0 = WMMA_BF16(ak1, bq1, s0);
      v16bf ak2 = load_frag_rm(Kb, kbase + 16 + ln, HD, 0,  hi);
      v16bf ak3 = load_frag_rm(Kb, kbase + 16 + ln, HD, 32, hi);
      s1 = WMMA_BF16(ak2, bq0, s1);
      s1 = WMMA_BF16(ak3, bq1, s1);
    }

    // Scale + causal mask + per-query (per-lane) running max.
    float mx = -INFINITY;
#pragma unroll
    for (int r = 0; r < 8; ++r) {
      const int key0 = kbase + r + hi * 8;
      const int key1 = key0 + 16;
      float v0 = s0[r] * scale; if (key0 > q) v0 = -INFINITY;
      float v1 = s1[r] * scale; if (key1 > q) v1 = -INFINITY;
      s0[r] = v0; s1[r] = v1;
      mx = fmaxf(mx, fmaxf(v0, v1));
    }
    mx = fmaxf(mx, __shfl_xor(mx, 16, 32));   // combine key halves
    const float m_new = fmaxf(m_st, mx);
    const float alpha = __expf(m_st - m_new); // 0 on first block

    // Probabilities -> directly the B-fragment for O^T = V^T * P.
    float rsum = 0.f;
    v16bf pb;
#pragma unroll
    for (int r = 0; r < 8; ++r) {
      const float p0 = __expf(s0[r] - m_new);
      const float p1 = __expf(s1[r] - m_new);
      rsum += p0 + p1;
      pb[r]     = (__bf16)p0;   // keys kbase + {0..7|8..15}
      pb[8 + r] = (__bf16)p1;   // keys kbase + {16..23|24..31}
    }
    rsum += __shfl_xor(rsum, 16, 32);
    l_st = l_st * alpha + rsum;
    m_st = m_new;

#pragma unroll
    for (int r = 0; r < 8; ++r) {
      o0[r] *= alpha; o1[r] *= alpha; o2[r] *= alpha; o3[r] *= alpha;
    }

    // V^T A-fragments: contiguous rows of VT[64, T].
    v16bf av0 = load_frag_rm(Vb,  0 + ln, T, kbase, hi);
    v16bf av1 = load_frag_rm(Vb, 16 + ln, T, kbase, hi);
    v16bf av2 = load_frag_rm(Vb, 32 + ln, T, kbase, hi);
    v16bf av3 = load_frag_rm(Vb, 48 + ln, T, kbase, hi);
    o0 = WMMA_BF16(av0, pb, o0);
    o1 = WMMA_BF16(av1, pb, o1);
    o2 = WMMA_BF16(av2, pb, o2);
    o3 = WMMA_BF16(av3, pb, o3);
  }

  // O^T layout: n = query = ln, m = dim = 16*dt + r + 8*hi.
  const float inv = 1.f / l_st;
  __bf16* dst = Aout + ((size_t)(b * T + q)) * D + h * HD;
  const int dbase = hi * 8;
#pragma unroll
  for (int r = 0; r < 8; ++r) {
    dst[ 0 + dbase + r] = (__bf16)(o0[r] * inv);
    dst[16 + dbase + r] = (__bf16)(o1[r] * inv);
    dst[32 + dbase + r] = (__bf16)(o2[r] * inv);
    dst[48 + dbase + r] = (__bf16)(o3[r] * inv);
  }
}

// ---------------------------------------------------------------------------
// Stage 3: output projection. out[m,n] = sum_k A[m,k]*Wout[n,k] + b[n] (f32)
// Same TDM-staged macro-tile structure as qkv_gemm.
// ---------------------------------------------------------------------------
__global__ __launch_bounds__(256) void out_gemm(const __bf16* __restrict__ A,
                                                const __bf16* __restrict__ W,
                                                const float* __restrict__ bias,
                                                float* __restrict__ out) {
  __shared__ __bf16 sA[128 * 64];
  __shared__ __bf16 sW[64 * 64];

  const int wave = threadIdx.x >> 5;
  const int lane = threadIdx.x & 31;
  const int hi   = lane >> 4;
  const int ln   = lane & 15;

  const int nblk = D / 64;                  // 16
  const int m0   = (blockIdx.x / nblk) * 128;
  const int n0   = (blockIdx.x % nblk) * 64;
  const int wm   = (wave >> 1) * 32;
  const int wn   = (wave & 1) * 32;

  const unsigned ldsA = (unsigned)(size_t)sA;
  const unsigned ldsW = (unsigned)(size_t)sW;

  v8f acc[2][2] = {};
  for (int k0 = 0; k0 < D; k0 += 64) {
    if (wave == 0) {
      tdm_load_2d_bf16(ldsA, A, (unsigned long long)m0 * D + k0,
                       D, Mrows, 64, 128, D);
      tdm_load_2d_bf16(ldsW, W, (unsigned long long)n0 * D + k0,
                       D, D, 64, 64, D);
      __builtin_amdgcn_s_wait_tensorcnt(0);
    }
    __syncthreads();
#pragma unroll
    for (int kk = 0; kk < 64; kk += 32) {
      v16bf a0 = load_frag_rm(sA, wm + ln,      64, kk, hi);
      v16bf a1 = load_frag_rm(sA, wm + 16 + ln, 64, kk, hi);
      v16bf b0 = load_frag_rm(sW, wn + ln,      64, kk, hi);
      v16bf b1 = load_frag_rm(sW, wn + 16 + ln, 64, kk, hi);
      acc[0][0] = WMMA_BF16(a0, b0, acc[0][0]);
      acc[0][1] = WMMA_BF16(a0, b1, acc[0][1]);
      acc[1][0] = WMMA_BF16(a1, b0, acc[1][0]);
      acc[1][1] = WMMA_BF16(a1, b1, acc[1][1]);
    }
    __syncthreads();
  }

#pragma unroll
  for (int tj = 0; tj < 2; ++tj) {
    const int n    = n0 + wn + tj * 16 + ln;
    const float bv = bias[n];
#pragma unroll
    for (int ti = 0; ti < 2; ++ti) {
#pragma unroll
      for (int r = 0; r < 8; ++r) {
        const int m = m0 + wm + ti * 16 + hi * 8 + r;
        out[(size_t)m * D + n] = acc[ti][tj][r] + bv;
      }
    }
  }
}

// ---------------------------------------------------------------------------
// Host launcher
// ---------------------------------------------------------------------------
extern "C" void kernel_launch(void* const* d_in, const int* in_sizes, int n_in,
                              void* d_out, int out_size, void* d_ws, size_t ws_size,
                              hipStream_t stream) {
  (void)in_sizes; (void)n_in; (void)out_size; (void)ws_size;

  const float* x     = (const float*)d_in[0];
  // d_in[1] = causal mask (bool) -- causality is computed analytically.
  const float* W_qkv = (const float*)d_in[2];
  const float* b_qkv = (const float*)d_in[3];
  const float* W_out = (const float*)d_in[4];
  const float* b_out = (const float*)d_in[5];
  float* out = (float*)d_out;

  // Workspace layout (bf16 buffers), total 48 MiB.
  char* ws = (char*)d_ws;
  __bf16* Xb    = (__bf16*)(ws);                              //  8 MiB: [4096,1024]
  __bf16* Wqkvb = (__bf16*)(ws + (8u  << 20));                //  6 MiB: [3072,1024]
  __bf16* Woutb = (__bf16*)(ws + (14u << 20));                //  2 MiB: [1024,1024]
  __bf16* Qm    = (__bf16*)(ws + (16u << 20));                //  8 MiB: [BH,T,64]
  __bf16* Km    = (__bf16*)(ws + (24u << 20));                //  8 MiB: [BH,T,64]
  __bf16* VTm   = (__bf16*)(ws + (32u << 20));                //  8 MiB: [BH,64,T]
  __bf16* Ao    = (__bf16*)(ws + (40u << 20));                //  8 MiB: [4096,1024]

  const int nX  = Mrows * D;       // 4194304
  const int nWq = Nqkv * D;        // 3145728
  const int nWo = D * D;           // 1048576

  cvt_f32_bf16<<<(nX  + 255) / 256, 256, 0, stream>>>(x,     Xb,    nX);
  cvt_f32_bf16<<<(nWq + 255) / 256, 256, 0, stream>>>(W_qkv, Wqkvb, nWq);
  cvt_f32_bf16<<<(nWo + 255) / 256, 256, 0, stream>>>(W_out, Woutb, nWo);

  // (4096/128)*(3072/64) = 32*48 = 1536 blocks
  qkv_gemm<<<(Mrows / 128) * (Nqkv / 64), 256, 0, stream>>>(
      Xb, Wqkvb, b_qkv, Qm, Km, VTm);

  // B*H*(T/128) = 512 blocks, each wave a 16-query tile
  attn_kernel<<<Bn * H * (T / 128), 256, 0, stream>>>(Qm, Km, VTm, Ao);

  // (4096/128)*(1024/64) = 32*16 = 512 blocks
  out_gemm<<<(Mrows / 128) * (D / 64), 256, 0, stream>>>(
      Ao, Woutb, b_out, out);
}
